// Attention_56272661512359
// MI455X (gfx1250) — matrix operand order repaired and checked by reading, not compiled, not run on hardware
//
#include <hip/hip_runtime.h>
#include <hip/hip_bf16.h>
#include <stdint.h>

#define BB 2
#define SS 2048
#define DDIM 2048
#define HH 16
#define HDD 128

typedef __bf16 v16bf __attribute__((ext_vector_type(16)));
typedef float  v8f   __attribute__((ext_vector_type(8)));
typedef unsigned int u32x4 __attribute__((ext_vector_type(4)));
typedef int          i32x8 __attribute__((ext_vector_type(8)));
typedef int          i32x4 __attribute__((ext_vector_type(4)));

union Frag { uint32_t u[8]; v16bf v; };

__device__ __forceinline__ uint16_t f2bf(float f) {
  uint32_t u = __float_as_uint(f);
  u += 0x7FFFu + ((u >> 16) & 1u);
  return (uint16_t)(u >> 16);
}
__device__ __forceinline__ float bf2f(uint16_t h) {
  return __uint_as_float(((uint32_t)h) << 16);
}

__device__ __forceinline__ v8f wmma_bf16(v16bf a, v16bf b, v8f c) {
  return __builtin_amdgcn_wmma_f32_16x16x32_bf16(false, a, false, b, (short)0, c,
                                                 false, false);
}

// ---------------------------------------------------------------- TDM 2-D tile load
// Builds a D# (groups 0/1 per CDNA5 ISA) for a 2-D bf16 tile:
//   tile_w elements per row (contiguous), tile_h rows, row stride = stride elems.
// Destination LDS region is contiguous row-major tile_h x tile_w.
__device__ __forceinline__ void tdm_load_2d_bf16(uint32_t lds_off,
                                                 const void* gptr,
                                                 uint32_t tile_w, uint32_t tile_h,
                                                 uint32_t tensor_w, uint32_t tensor_h,
                                                 uint32_t stride) {
  uint64_t ga = (uint64_t)(uintptr_t)gptr;
  u32x4 g0;
  g0[0] = 1u;                                     // count=1, user descriptor
  g0[1] = lds_off;                                // lds_addr
  g0[2] = (uint32_t)(ga & 0xFFFFFFFFu);           // global_addr[31:0]
  g0[3] = (uint32_t)((ga >> 32) & 0x01FFFFFFu)    // global_addr[56:32]
          | 0x80000000u;                          // type=2 ("image")
  i32x8 g1;
  g1[0] = 0x00010000;                             // data_size=1 (2 bytes)
  g1[1] = (int)((tensor_w & 0xFFFFu) << 16);      // tensor_dim0[15:0]
  g1[2] = (int)((tensor_w >> 16) & 0xFFFFu)       // tensor_dim0[31:16]
          | (int)((tensor_h & 0xFFFFu) << 16);    // tensor_dim1[15:0]
  g1[3] = (int)((tensor_h >> 16) & 0xFFFFu)       // tensor_dim1[31:16]
          | (int)(tile_w << 16);                  // tile_dim0
  g1[4] = (int)(tile_h & 0xFFFFu);                // tile_dim1 (tile_dim2=0)
  g1[5] = (int)stride;                            // tensor_dim0_stride[31:0]
  g1[6] = 0;                                      // stride[47:32], dim1_stride[15:0]
  g1[7] = 0;
  i32x4 z4 = {0, 0, 0, 0};
  i32x8 z8 = {0, 0, 0, 0, 0, 0, 0, 0};
  __builtin_amdgcn_tensor_load_to_lds(g0, g1, z4, z4, z8, 0);
}

__device__ __forceinline__ uint32_t lds_offset(const void* p) {
  return (uint32_t)(uintptr_t)p;  // flat LDS addr: addr[31:0] is the LDS offset
}

// A fragment: 16x32 (MxK), element (m,k) at base[m*rs + k]; bf16 pairs (k even)
__device__ __forceinline__ v16bf load_frag_a(const uint16_t* base, int rs) {
  int lane = (int)(threadIdx.x & 31u);
  int row = lane & 15, g = lane >> 4;
  Frag f;
#pragma unroll
  for (int v = 0; v < 8; ++v) {
    int k = (g << 4) + (v << 1);
    f.u[v] = *(const uint32_t*)(base + row * rs + k);
  }
  return f.v;
}

// B fragment: 32x16 (KxN), element (k,n) at base[n*rs + k] (k contiguous).
__device__ __forceinline__ v16bf load_frag_b(const uint16_t* base, int rs) {
  int lane = (int)(threadIdx.x & 31u);
  int col = lane & 15, g = lane >> 4;
  Frag f;
#pragma unroll
  for (int v = 0; v < 8; ++v) {
    int k = (v << 1) + (g << 4);
    f.u[v] = *(const uint32_t*)(base + col * rs + k);
  }
  return f.v;
}

// B fragment: 32x16 (KxN), element (k,n) at base[k*rs + n] (k strided).
__device__ __forceinline__ v16bf load_frag_b_strided(const uint16_t* base, int rs) {
  int lane = (int)(threadIdx.x & 31u);
  int col = lane & 15, g = lane >> 4;
  Frag f;
#pragma unroll
  for (int v = 0; v < 8; ++v) {
    int k = (v << 1) + (g << 4);
    uint32_t lo = base[(size_t)k * rs + col];
    uint32_t hi = base[(size_t)(k + 1) * rs + col];
    f.u[v] = lo | (hi << 16);
  }
  return f.v;
}

// ---------------------------------------------------------------- cast f32->bf16
__global__ __launch_bounds__(256) void cast_f32_bf16(const float* __restrict__ in,
                                                     uint16_t* __restrict__ out,
                                                     int n) {
  int i = (int)(blockIdx.x * 256 + threadIdx.x);
  if (i < n) out[i] = f2bf(in[i]);
}

// ------------------------------------------------- C[M,N] = A[M,K] * B[N,K]^T
// 128x128 block tile, K-step 32, 256 threads (8 waves), wave tile 32x64.
// TDM double-buffered: wave 0 DMAs tile i+1 while all waves run WMMA on tile i.
template <bool F32OUT>
__global__ __launch_bounds__(256) void gemm_xwT(const uint16_t* __restrict__ A,
                                                const uint16_t* __restrict__ Bw,
                                                void* __restrict__ C,
                                                int M, int N, int K) {
  __shared__ __align__(16) uint16_t As[2][128 * 32];
  __shared__ __align__(16) uint16_t Bs[2][128 * 32];

  const int bm = blockIdx.y * 128;
  const int bn = blockIdx.x * 128;
  const int tid = (int)threadIdx.x;
  const int wave = tid >> 5;
  const int wm = (wave & 3) * 32;   // 2 m-tiles
  const int wn = (wave >> 2) * 64;  // 4 n-tiles

  v8f acc[2][4];
#pragma unroll
  for (int mi = 0; mi < 2; ++mi)
#pragma unroll
    for (int ni = 0; ni < 4; ++ni) acc[mi][ni] = (v8f)0.0f;

  const int nk = K / 32;
  if (wave == 0) {
    tdm_load_2d_bf16(lds_offset(&As[0][0]), A + (size_t)bm * K, 32, 128,
                     (uint32_t)K, 128, (uint32_t)K);
    tdm_load_2d_bf16(lds_offset(&Bs[0][0]), Bw + (size_t)bn * K, 32, 128,
                     (uint32_t)K, 128, (uint32_t)K);
    __builtin_amdgcn_s_wait_tensorcnt((short)0);
  }
  __syncthreads();

  for (int ik = 0; ik < nk; ++ik) {
    const int cur = ik & 1;
    if (wave == 0 && (ik + 1) < nk) {
      const int k0 = (ik + 1) * 32;
      tdm_load_2d_bf16(lds_offset(&As[cur ^ 1][0]), A + (size_t)bm * K + k0, 32, 128,
                       (uint32_t)K, 128, (uint32_t)K);
      tdm_load_2d_bf16(lds_offset(&Bs[cur ^ 1][0]), Bw + (size_t)bn * K + k0, 32, 128,
                       (uint32_t)K, 128, (uint32_t)K);
    }

    v16bf af[2], bf[4];
#pragma unroll
    for (int mi = 0; mi < 2; ++mi)
      af[mi] = load_frag_a(&As[cur][(wm + mi * 16) * 32], 32);
#pragma unroll
    for (int ni = 0; ni < 4; ++ni)
      bf[ni] = load_frag_b(&Bs[cur][(wn + ni * 16) * 32], 32);
#pragma unroll
    for (int mi = 0; mi < 2; ++mi)
#pragma unroll
      for (int ni = 0; ni < 4; ++ni)
        acc[mi][ni] = wmma_bf16(af[mi], bf[ni], acc[mi][ni]);

    if (wave == 0) __builtin_amdgcn_s_wait_tensorcnt((short)0);
    __syncthreads();
  }

  const int lane = tid & 31, g = lane >> 4, col = lane & 15;
#pragma unroll
  for (int mi = 0; mi < 2; ++mi)
#pragma unroll
    for (int ni = 0; ni < 4; ++ni)
#pragma unroll
      for (int r = 0; r < 8; ++r) {
        int row = bm + wm + mi * 16 + r + 8 * g;
        int cc = bn + wn + ni * 16 + col;
        float v = acc[mi][ni][r];
        if (F32OUT)
          ((float*)C)[(size_t)row * N + cc] = v;
        else
          ((uint16_t*)C)[(size_t)row * N + cc] = f2bf(v);
      }
}

// ------------------------------------------------------------------- RoPE (in place)
__global__ __launch_bounds__(256) void rope_kernel(uint16_t* __restrict__ t,
                                                   const float* __restrict__ cosp,
                                                   const float* __restrict__ sinp) {
  int idx = (int)(blockIdx.x * 256 + threadIdx.x);  // pair index
  const int total = BB * SS * HH * (HDD / 2);
  if (idx >= total) return;
  int i = idx & 63;           // HD/2 = 64
  int h = (idx >> 6) & 15;    // H = 16
  int bs = idx >> 10;         // b*S + s
  int s = bs & (SS - 1);
  size_t off = (size_t)bs * DDIM + (size_t)h * HDD + 2 * i;
  float c = cosp[s * 64 + i], sn = sinp[s * 64 + i];
  float e = bf2f(t[off]), o = bf2f(t[off + 1]);
  t[off] = f2bf(e * c - o * sn);
  t[off + 1] = f2bf(e * sn + o * c);
}

// ---------------------------------------------------- flash attention, causal
// grid: (S/64, H, B); block: 128 threads = 4 waves, wave owns 16 query rows.
// K/V tiles filled by TDM (wave 0), tracked with TENSORcnt.
__global__ __launch_bounds__(128) void flash_attn(const uint16_t* __restrict__ Qb,
                                                  const uint16_t* __restrict__ Kb,
                                                  const uint16_t* __restrict__ Vb,
                                                  uint16_t* __restrict__ Ob) {
  __shared__ __align__(16) uint16_t Qt[64 * 128];
  __shared__ __align__(16) uint16_t Kt[64 * 128];
  __shared__ __align__(16) uint16_t Vt[64 * 128];
  __shared__ __align__(16) uint16_t Pst[4][16 * 64];

  const int qb = (int)blockIdx.x;
  const int h = (int)blockIdx.y;
  const int b = (int)blockIdx.z;
  const int tid = (int)threadIdx.x;
  const int wave = tid >> 5;
  const int lane = tid & 31, g = lane >> 4, ln = lane & 15;
  const float scale = 0.08838834764831845f;  // 1/sqrt(128)

  const size_t baseQ = ((size_t)b * SS + qb * 64) * DDIM + (size_t)h * HDD;
#pragma unroll
  for (int i = 0; i < 8; ++i) {
    int idx = i * 1024 + tid * 8;
    int r = idx >> 7, c = idx & 127;
    *(uint4*)(&Qt[idx]) = *(const uint4*)(Qb + baseQ + (size_t)r * DDIM + c);
  }
  __syncthreads();

  v16bf qf[4];
#pragma unroll
  for (int c = 0; c < 4; ++c) qf[c] = load_frag_a(&Qt[wave * 16 * 128 + c * 32], 128);

  float mrow[8], lrow[8];
  v8f acc[8];
#pragma unroll
  for (int r = 0; r < 8; ++r) { mrow[r] = -3.0e38f; lrow[r] = 0.0f; }
#pragma unroll
  for (int d = 0; d < 8; ++d) acc[d] = (v8f)0.0f;

  for (int kb = 0; kb <= qb; ++kb) {
    __syncthreads();  // all waves done with previous Kt/Vt
    const size_t baseK = ((size_t)b * SS + kb * 64) * DDIM + (size_t)h * HDD;
    if (wave == 0) {
      tdm_load_2d_bf16(lds_offset(&Kt[0]), Kb + baseK, 128, 64,
                       (uint32_t)DDIM, 64, (uint32_t)DDIM);
      tdm_load_2d_bf16(lds_offset(&Vt[0]), Vb + baseK, 128, 64,
                       (uint32_t)DDIM, 64, (uint32_t)DDIM);
      __builtin_amdgcn_s_wait_tensorcnt((short)0);
    }
    __syncthreads();

    // S = Q K^T  (wave's 16 rows x 64 keys)
    v8f sf[4];
#pragma unroll
    for (int nt = 0; nt < 4; ++nt) {
      v8f s = (v8f)0.0f;
#pragma unroll
      for (int c = 0; c < 4; ++c)
        s = wmma_bf16(qf[c], load_frag_b(&Kt[nt * 16 * 128 + c * 32], 128), s);
      sf[nt] = s;
    }

    // scale (+ causal mask on diagonal block)
#pragma unroll
    for (int nt = 0; nt < 4; ++nt)
#pragma unroll
      for (int r = 0; r < 8; ++r) {
        float v = sf[nt][r] * scale;
        if (kb == qb) {
          int qrow = wave * 16 + r + 8 * g;
          int krow = nt * 16 + ln;
          if (krow > qrow) v = -1.0e30f;
        }
        sf[nt][r] = v;
      }

    // online softmax: row max / row sum across the 16 lanes sharing each row
    float mnew[8], alpha[8];
#pragma unroll
    for (int r = 0; r < 8; ++r) {
      float mx = fmaxf(fmaxf(sf[0][r], sf[1][r]), fmaxf(sf[2][r], sf[3][r]));
#pragma unroll
      for (int off = 1; off < 16; off <<= 1) mx = fmaxf(mx, __shfl_xor(mx, off, 32));
      mnew[r] = fmaxf(mrow[r], mx);
      alpha[r] = __expf(mrow[r] - mnew[r]);
      mrow[r] = mnew[r];
    }
#pragma unroll
    for (int nt = 0; nt < 4; ++nt)
#pragma unroll
      for (int r = 0; r < 8; ++r) sf[nt][r] = __expf(sf[nt][r] - mnew[r]);
#pragma unroll
    for (int r = 0; r < 8; ++r) {
      float sum = sf[0][r] + sf[1][r] + sf[2][r] + sf[3][r];
#pragma unroll
      for (int off = 1; off < 16; off <<= 1) sum += __shfl_xor(sum, off, 32);
      lrow[r] = lrow[r] * alpha[r] + sum;
    }
#pragma unroll
    for (int d = 0; d < 8; ++d)
#pragma unroll
      for (int r = 0; r < 8; ++r) acc[d][r] *= alpha[r];

    // stage P (C layout -> LDS row-major) for re-load as A fragments
#pragma unroll
    for (int nt = 0; nt < 4; ++nt)
#pragma unroll
      for (int r = 0; r < 8; ++r)
        Pst[wave][(r + 8 * g) * 64 + nt * 16 + ln] = f2bf(sf[nt][r]);
    __syncthreads();

    v16bf pf[2];
#pragma unroll
    for (int c = 0; c < 2; ++c) pf[c] = load_frag_a(&Pst[wave][c * 32], 64);
#pragma unroll
    for (int dt = 0; dt < 8; ++dt)
#pragma unroll
      for (int c = 0; c < 2; ++c)
        acc[dt] = wmma_bf16(pf[c],
                            load_frag_b_strided(&Vt[c * 32 * 128 + dt * 16], 128),
                            acc[dt]);
  }

  const size_t baseO = ((size_t)b * SS + qb * 64 + wave * 16) * DDIM + (size_t)h * HDD;
#pragma unroll
  for (int dt = 0; dt < 8; ++dt)
#pragma unroll
    for (int r = 0; r < 8; ++r) {
      int row = r + 8 * g;
      int col = dt * 16 + ln;
      float o = acc[dt][r] / lrow[r];
      Ob[baseO + (size_t)row * DDIM + col] = f2bf(o);
    }
}

// ---------------------------------------------------------------------- launch
extern "C" void kernel_launch(void* const* d_in, const int* in_sizes, int n_in,
                              void* d_out, int out_size, void* d_ws, size_t ws_size,
                              hipStream_t stream) {
  (void)in_sizes; (void)n_in; (void)out_size; (void)ws_size;
  const float* x  = (const float*)d_in[0];
  const float* fc = (const float*)d_in[1];
  const float* fs = (const float*)d_in[2];
  // d_in[3] = mask (unused; causal mask computed analytically)
  const float* wq = (const float*)d_in[4];
  const float* wk = (const float*)d_in[5];
  const float* wv = (const float*)d_in[6];
  const float* wo = (const float*)d_in[7];

  const size_t NX = (size_t)BB * SS * DDIM;  // 8388608
  const size_t NW = (size_t)DDIM * DDIM;     // 4194304

  uint16_t* ws  = (uint16_t*)d_ws;
  uint16_t* xb  = ws;
  uint16_t* wqb = xb + NX;
  uint16_t* wkb = wqb + NW;
  uint16_t* wvb = wkb + NW;
  uint16_t* wob = wvb + NW;
  uint16_t* Qb  = wob + NW;
  uint16_t* Kb  = Qb + NX;
  uint16_t* Vb  = Kb + NX;
  uint16_t* Ob  = Vb + NX;

  cast_f32_bf16<<<(unsigned)((NX + 255) / 256), 256, 0, stream>>>(x, xb, (int)NX);
  cast_f32_bf16<<<(unsigned)((NW + 255) / 256), 256, 0, stream>>>(wq, wqb, (int)NW);
  cast_f32_bf16<<<(unsigned)((NW + 255) / 256), 256, 0, stream>>>(wk, wkb, (int)NW);
  cast_f32_bf16<<<(unsigned)((NW + 255) / 256), 256, 0, stream>>>(wv, wvb, (int)NW);
  cast_f32_bf16<<<(unsigned)((NW + 255) / 256), 256, 0, stream>>>(wo, wob, (int)NW);

  dim3 gg(DDIM / 128, (BB * SS) / 128);  // (16, 32)
  gemm_xwT<false><<<gg, 256, 0, stream>>>(xb, wqb, Qb, BB * SS, DDIM, DDIM);
  gemm_xwT<false><<<gg, 256, 0, stream>>>(xb, wkb, Kb, BB * SS, DDIM, DDIM);
  gemm_xwT<false><<<gg, 256, 0, stream>>>(xb, wvb, Vb, BB * SS, DDIM, DDIM);

  const int pairs = BB * SS * HH * (HDD / 2);
  rope_kernel<<<pairs / 256, 256, 0, stream>>>(Qb, fc, fs);
  rope_kernel<<<pairs / 256, 256, 0, stream>>>(Kb, fc, fs);

  flash_attn<<<dim3(SS / 64, HH, BB), 128, 0, stream>>>(Qb, Kb, Vb, Ob);

  gemm_xwT<true><<<gg, 256, 0, stream>>>(Ob, wob, (float*)d_out, BB * SS, DDIM, DDIM);
}
